// ABCAttention_50740743635382
// MI455X (gfx1250) — compile-verified, matrix operand order checked by reading, not run on hardware
//
#include <hip/hip_runtime.h>
#include <hip/hip_bf16.h>

#define B_  4
#define T_  2048
#define D_  1024
#define H_  4
#define DK_ 256
#define DV_ 256
#define M_  256
#define N5_ 5120   // q|k|v|g|s projection width
#define NW_ 6144   // packed weight width (adds o_w)

typedef __attribute__((ext_vector_type(16))) _Float16 v16h;
typedef __attribute__((ext_vector_type(8)))  float    v8f;
typedef __attribute__((ext_vector_type(4)))  unsigned int u32x4;
typedef __attribute__((ext_vector_type(8)))  int          i32x8;
typedef __attribute__((ext_vector_type(4)))  int          i32x4;

static __device__ __forceinline__ v8f wmma_f16(v16h a, v16h b, v8f c) {
  // 8 args: (neg_a, A, neg_b, B, c_mod, C, reuse_a, reuse_b)
  return __builtin_amdgcn_wmma_f32_16x16x32_f16(false, a, false, b, (short)0, c,
                                                false, false);
}

// 16x32 f16 operand fragment from a row-major source where lane&15 selects the
// "row" (A: M row, B: N column of a transposed tile) and K pairs follow the ISA
// 16-bit striping (lanes<16: K 0-7,16-23; lanes>=16: +8). Per lane this is two
// contiguous 16-byte chunks -> lowers to ds_load_b128/global_load_b128 pairs.
static __device__ __forceinline__ v16h load_frag_rowmajor(const _Float16* base,
                                                          int ld, int lane) {
  const int r    = lane & 15;
  const int koff = (lane >> 4) << 3;
  v16h f;
#pragma unroll
  for (int p = 0; p < 8; ++p) {
    const int kb = ((p & 4) << 2) + ((p & 3) << 1) + koff;
    f[2 * p]     = base[(size_t)r * ld + kb];
    f[2 * p + 1] = base[(size_t)r * ld + kb + 1];
  }
  return f;
}

// TDM: DMA a 2D tile (rows x 32 halfs) from global (row stride = lda halfs)
// into LDS with 8-half padding after each 32-half row (40-half pitch).
// Issued by one lane; EXEC is ignored for tensor ops; completion via TENSORcnt.
static __device__ __forceinline__ void tdm_load_tile_32h(
    const _Float16* gptr, unsigned lds_off, int rows, int lda,
    int tensor_rows) {
  const unsigned long long ga = (unsigned long long)(size_t)gptr;
  u32x4 g0;
  g0[0] = 1u;                                   // count=1 (valid), user mode
  g0[1] = lds_off;                              // lds_addr (bytes)
  g0[2] = (unsigned)ga;                         // global_addr[31:0]
  g0[3] = (unsigned)(ga >> 32) | (2u << 30);    // global_addr[56:32], type=2
  i32x8 g1;
  // data_size=1 (2B) @16 | pad_enable @20 | pad_interval=3 (16 DW) @22 |
  // pad_amount=3 (4 DW) @25
  g1[0] = (1 << 16) | (1 << 20) | (3 << 22) | (3 << 25);
  g1[1] = (int)(((unsigned)lda & 0xFFFFu) << 16);           // tensor_dim0 lo16
  g1[2] = (int)((((unsigned)lda >> 16) & 0xFFFFu) |
                (((unsigned)tensor_rows & 0xFFFFu) << 16)); // dim0 hi | dim1 lo
  g1[3] = (int)((((unsigned)tensor_rows >> 16) & 0xFFFFu) |
                (32u << 16));                               // dim1 hi | tile_dim0
  g1[4] = (int)((unsigned)rows & 0xFFFFu);                  // tile_dim1
  g1[5] = lda;                                              // dim0_stride lo32
  g1[6] = 0;                                                // stride hi, dim1_stride lo
  g1[7] = 0;
  const i32x4 gz = {0, 0, 0, 0};
#if __clang_major__ >= 23
  const i32x8 gz8 = {0, 0, 0, 0, 0, 0, 0, 0};
  __builtin_amdgcn_tensor_load_to_lds(g0, g1, gz, gz, gz8, 0);
#else
  __builtin_amdgcn_tensor_load_to_lds(g0, g1, gz, gz, 0);
#endif
}

// ---------------------------------------------------------------- casts
__global__ __launch_bounds__(256) void cast_x_kernel(const float* __restrict__ x,
                                                     _Float16* __restrict__ xh) {
  const size_t i = (size_t)blockIdx.x * blockDim.x + threadIdx.x;
  if (i < (size_t)B_ * T_ * D_) xh[i] = (_Float16)x[i];
}

__global__ __launch_bounds__(256) void cast_w_kernel(
    const float* __restrict__ qw, const float* __restrict__ kw,
    const float* __restrict__ vw, const float* __restrict__ gw,
    const float* __restrict__ sw, const float* __restrict__ ow,
    _Float16* __restrict__ wh) {
  const size_t i = (size_t)blockIdx.x * blockDim.x + threadIdx.x;
  if (i >= (size_t)D_ * NW_) return;
  const int n = (int)(i % NW_);
  const int k = (int)(i / NW_);
  float v;
  if (n < 1024)      v = qw[(size_t)k * D_ + n];
  else if (n < 2048) v = kw[(size_t)k * D_ + n - 1024];
  else if (n < 3072) v = vw[(size_t)k * D_ + n - 2048];
  else if (n < 4096) v = gw[(size_t)k * D_ + n - 3072];
  else if (n < 5120) v = sw[(size_t)k * D_ + n - 4096];
  else               v = ow[(size_t)k * D_ + n - 5120];
  wh[i] = (_Float16)v;
}

// ------------------------------------------------- tiled WMMA GEMM (f16->f32)
// C[M x N] = A[M x K] * B[K x N]; block tile 128x128, K-step 32, 8 waves.
// A tile staged by the Tensor Data Mover (async DMA + LDS padding); B tile
// staged transposed by the VALU so both operands load as ds_load_b128 pairs.
__global__ __launch_bounds__(256) void gemm_kernel(
    const _Float16* __restrict__ A, const _Float16* __restrict__ Bw,
    float* __restrict__ C, int K, int ldb, int ldc, int mrows) {
  __shared__ _Float16 sA[128][40];    // [m][k] (40-half pitch = TDM pad)
  __shared__ _Float16 sBT[128][40];   // [n][k]
  const int tid  = threadIdx.x;
  const int lane = tid & 31;
  const int wave = tid >> 5;
  const int bm   = blockIdx.x * 128;
  const int bn   = blockIdx.y * 128;
  const int wm   = (wave & 1) * 64;   // 2 wave-rows of 64
  const int wn   = (wave >> 1) * 32;  // 4 wave-cols of 32
  const unsigned lds_a = (unsigned)(size_t)&sA[0][0];

  const v8f vzero = {0.f, 0.f, 0.f, 0.f, 0.f, 0.f, 0.f, 0.f};
  v8f acc[4][2];
#pragma unroll
  for (int i = 0; i < 4; ++i)
#pragma unroll
    for (int j = 0; j < 2; ++j) acc[i][j] = vzero;

  for (int k0 = 0; k0 < K; k0 += 32) {
    if (tid == 0) {  // A tile: 128 rows x 32 halfs via TDM
      tdm_load_tile_32h(&A[(size_t)bm * K + k0], lds_a, 128, K, mrows);
    }
#pragma unroll
    for (int u = 0; u < 2; ++u) {  // B tile 32x128, vector read + transpose
      const int g  = tid + u * 256;    // 0..511
      const int k  = g >> 4;           // 0..31
      const int n0 = (g & 15) * 8;
      const uint4 d = *reinterpret_cast<const uint4*>(
          &Bw[(size_t)(k0 + k) * ldb + bn + n0]);
      const _Float16* hp = reinterpret_cast<const _Float16*>(&d);
#pragma unroll
      for (int j = 0; j < 8; ++j) sBT[n0 + j][k] = hp[j];
    }
    __builtin_amdgcn_s_wait_tensorcnt(0);
    __syncthreads();
    if (k0 + 32 < K) {  // pull next B tile toward L2 (global_prefetch)
      __builtin_prefetch(&Bw[(size_t)(k0 + 32 + (tid >> 3)) * ldb + bn], 0, 1);
    }
    v16h bfr[2];
#pragma unroll
    for (int j = 0; j < 2; ++j)
      bfr[j] = load_frag_rowmajor(&sBT[wn + j * 16][0], 40, lane);
#pragma unroll
    for (int i = 0; i < 4; ++i) {
      v16h af = load_frag_rowmajor(&sA[wm + i * 16][0], 40, lane);
#pragma unroll
      for (int j = 0; j < 2; ++j) acc[i][j] = wmma_f16(af, bfr[j], acc[i][j]);
    }
    __syncthreads();
  }
  const int col  = lane & 15;
  const int rofs = (lane >> 4) << 3;
#pragma unroll
  for (int i = 0; i < 4; ++i)
#pragma unroll
    for (int j = 0; j < 2; ++j)
#pragma unroll
      for (int v = 0; v < 8; ++v)
        C[(size_t)(bm + wm + i * 16 + rofs + v) * ldc + bn + wn + j * 16 + col] =
            acc[i][j][v];
}

// ----------------------------------------- sg = log_sigmoid(x @ sg_w)/16
__global__ __launch_bounds__(256) void sg_kernel(const float* __restrict__ x,
                                                 const float* __restrict__ sgw,
                                                 float* __restrict__ sgbuf) {
  const int idx = blockIdx.x * blockDim.x + threadIdx.x;  // (b*T+t)*H + h
  if (idx >= B_ * T_ * H_) return;
  const int h  = idx & 3;
  const int bt = idx >> 2;
  const float* xr = x + (size_t)bt * D_;
  float acc = 0.f;
  for (int k = 0; k < D_; ++k) acc += xr[k] * sgw[k * H_ + h];
  const float ls = fminf(acc, 0.f) - log1pf(__expf(-fabsf(acc)));
  const int b = bt / T_, t = bt % T_;
  sgbuf[((size_t)(b * H_ + h)) * T_ + t] = ls * (1.0f / 16.0f);
}

__global__ void cumsum_kernel(float* __restrict__ sgbuf) {  // in-place -> decay
  const int bh = threadIdx.x;
  if (bh >= B_ * H_) return;
  float a = 0.f;
  for (int t = 0; t < T_; ++t) {
    a += sgbuf[(size_t)bh * T_ + t];
    sgbuf[(size_t)bh * T_ + t] = a;
  }
}

// ---------------------------------- RoPE on q,k ; k *= exp(cumsum(sg))
__global__ __launch_bounds__(256) void rope_kernel(const float* __restrict__ Y5,
                                                   const float* __restrict__ dec,
                                                   _Float16* __restrict__ qh,
                                                   _Float16* __restrict__ kh) {
  const size_t idx = (size_t)blockIdx.x * blockDim.x + threadIdx.x;
  if (idx >= (size_t)B_ * H_ * T_ * 128) return;
  const int i  = (int)(idx & 127);
  const size_t rt = idx >> 7;          // bh*T + t
  const int t  = (int)(rt % T_);
  const int bh = (int)(rt / T_);
  const int h  = bh & 3, b = bh >> 2;
  const float inv = __powf(10000.0f, -(float)i / 128.0f);
  float s, c;
  __sincosf((float)t * inv, &s, &c);
  const size_t src = (size_t)(b * T_ + t) * N5_;
  const float q1 = Y5[src + 0 * D_ + h * DK_ + i];
  const float q2 = Y5[src + 0 * D_ + h * DK_ + i + 128];
  const float k1 = Y5[src + 1 * D_ + h * DK_ + i];
  const float k2 = Y5[src + 1 * D_ + h * DK_ + i + 128];
  const float dk = __expf(dec[(size_t)bh * T_ + t]);
  const size_t dst = ((size_t)bh * T_ + t) * DK_;
  qh[dst + i]       = (_Float16)(q1 * c - q2 * s);
  qh[dst + i + 128] = (_Float16)(q2 * c + q1 * s);
  kh[dst + i]       = (_Float16)((k1 * c - k2 * s) * dk);
  kh[dst + i + 128] = (_Float16)((k2 * c + k1 * s) * dk);
}

// ------------------------- cumlogsumexp over t of clamped s; es = exp(s-32)
__global__ __launch_bounds__(256) void zscan_kernel(const float* __restrict__ Y5,
                                                    float* __restrict__ zbuf,
                                                    _Float16* __restrict__ es) {
  const int idx = blockIdx.x * blockDim.x + threadIdx.x;  // bh*M + m
  if (idx >= B_ * H_ * M_) return;
  const int m  = idx & (M_ - 1);
  const int bh = idx >> 8;
  const int h  = bh & 3, b = bh >> 2;
  float zr = -__builtin_inff();
  for (int t = 0; t < T_; ++t) {
    float s = Y5[(size_t)(b * T_ + t) * N5_ + 4 * D_ + h * M_ + m];
    s = fminf(fmaxf(s, -32.f), 32.f);
    const float mx = fmaxf(zr, s);
    zr = mx + logf(__expf(zr - mx) + __expf(s - mx));
    const size_t o = ((size_t)bh * T_ + t) * M_ + m;
    zbuf[o] = zr;
    es[o]   = (_Float16)__expf(s - 32.0f);
  }
}

__global__ __launch_bounds__(256) void vcast_kernel(const float* __restrict__ Y5,
                                                    _Float16* __restrict__ vh) {
  const size_t idx = (size_t)blockIdx.x * blockDim.x + threadIdx.x;
  if (idx >= (size_t)B_ * H_ * T_ * DV_) return;
  const int dv = (int)(idx & 255);
  const size_t rt = idx >> 8;
  const int t = (int)(rt % T_), bh = (int)(rt / T_);
  const int h = bh & 3, b = bh >> 2;
  vh[idx] = (_Float16)Y5[(size_t)(b * T_ + t) * N5_ + 2 * D_ + h * DV_ + dv];
}

// ------------- stage A: ok[t,m] = exp(32-z) * sum_{tau<=t} (q.k/16) es[tau,m]
__global__ __launch_bounds__(256) void attn1_kernel(
    const _Float16* __restrict__ qh, const _Float16* __restrict__ kh,
    const _Float16* __restrict__ es, const float* __restrict__ zbuf,
    float* __restrict__ okbuf) {
  const int qt   = blockIdx.x;   // 16-row query tile
  const int bh   = blockIdx.y;
  const int lane = threadIdx.x & 31;
  const int wave = threadIdx.x >> 5;
  __shared__ _Float16 ldsA[16][40];    // masked score tile [t][tau]
  __shared__ _Float16 sET[256][40];    // ES chunk transposed [m][tau]

  const _Float16* qbase = qh + ((size_t)bh * T_ + qt * 16) * DK_;
  const _Float16* kbase = kh + (size_t)bh * T_ * DK_;
  const _Float16* ebase = es + (size_t)bh * T_ * M_;
  const int m0 = wave * 32;

  v16h aq[8];
  if (wave < 2) {
#pragma unroll
    for (int kc = 0; kc < 8; ++kc)
      aq[kc] = load_frag_rowmajor(qbase + kc * 32, DK_, lane);
  }

  const v8f vzero = {0.f, 0.f, 0.f, 0.f, 0.f, 0.f, 0.f, 0.f};
  v8f accY[2] = {vzero, vzero};

  const int nchunks = (qt >> 1) + 1;
  for (int c = 0; c < nchunks; ++c) {
    const int t0 = c * 32;
    // stage ES chunk (32 tau x 256 m) transposed: uint4 reads, b16 scatter
#pragma unroll
    for (int u = 0; u < 4; ++u) {
      const int g   = threadIdx.x + u * 256;  // 0..1023
      const int tau = g >> 5;                 // 0..31
      const int mg  = (g & 31) * 8;
      const uint4 d = *reinterpret_cast<const uint4*>(
          &ebase[(size_t)(t0 + tau) * M_ + mg]);
      const _Float16* hp = reinterpret_cast<const _Float16*>(&d);
#pragma unroll
      for (int j = 0; j < 8; ++j) sET[mg + j][tau] = hp[j];
    }
    if (wave < 2) {  // 16x16 score sub-tile, contraction over DK
      v8f accA = vzero;
      const _Float16* kb = kbase + (size_t)(t0 + wave * 16) * DK_;
#pragma unroll
      for (int kc = 0; kc < 8; ++kc) {
        v16h bf = load_frag_rowmajor(kb + kc * 32, DK_, lane);  // K^T gather
        accA = wmma_f16(aq[kc], bf, accA);
      }
      const int col  = lane & 15;
      const int rofs = (lane >> 4) << 3;
      const int taug = t0 + wave * 16 + col;
#pragma unroll
      for (int i = 0; i < 8; ++i) {
        const int row = i + rofs;
        const int tg  = qt * 16 + row;
        ldsA[row][wave * 16 + col] =
            (_Float16)((taug <= tg) ? accA[i] * 0.0625f : 0.0f);
      }
    }
    __syncthreads();
    v16h af = load_frag_rowmajor(&ldsA[0][0], 40, lane);
#pragma unroll
    for (int s = 0; s < 2; ++s) {
      v16h bf = load_frag_rowmajor(&sET[m0 + s * 16][0], 40, lane);
      accY[s] = wmma_f16(af, bf, accY[s]);
    }
    __syncthreads();
  }
  const int col  = lane & 15;
  const int rofs = (lane >> 4) << 3;
#pragma unroll
  for (int s = 0; s < 2; ++s) {
    const int m = m0 + s * 16 + col;
#pragma unroll
    for (int i = 0; i < 8; ++i) {
      const int t = qt * 16 + i + rofs;
      const size_t o = ((size_t)bh * T_ + t) * M_ + m;
      okbuf[o] = accY[s][i] * __expf(32.0f - zbuf[o]);
    }
  }
}

// -------- softmax over m, qz = qv*exp(zmin-z) (f16), rowscale = exp(32-zmin)
__global__ __launch_bounds__(256) void softmax_qz_kernel(
    const float* __restrict__ okbuf, const float* __restrict__ zbuf,
    _Float16* __restrict__ qzh, float* __restrict__ rsc) {
  const int lane = threadIdx.x & 31;
  const int wave = threadIdx.x >> 5;
  const size_t row = (size_t)blockIdx.x * 8 + wave;  // bh*T + t
  const float* ok = okbuf + row * M_;
  const float* z  = zbuf + row * M_;
  float vals[8], zl[8];
  float mx = -__builtin_inff(), zmin = __builtin_inff();
#pragma unroll
  for (int i = 0; i < 8; ++i) {
    vals[i] = ok[lane + i * 32];
    zl[i]   = z[lane + i * 32];
    mx   = fmaxf(mx, vals[i]);
    zmin = fminf(zmin, zl[i]);
  }
#pragma unroll
  for (int off = 16; off; off >>= 1) {
    mx   = fmaxf(mx, __shfl_xor(mx, off, 32));
    zmin = fminf(zmin, __shfl_xor(zmin, off, 32));
  }
  float sum = 0.f;
#pragma unroll
  for (int i = 0; i < 8; ++i) {
    vals[i] = __expf(vals[i] - mx);
    sum += vals[i];
  }
#pragma unroll
  for (int off = 16; off; off >>= 1) sum += __shfl_xor(sum, off, 32);
  const float inv = 1.0f / sum;
#pragma unroll
  for (int i = 0; i < 8; ++i)
    qzh[row * M_ + lane + i * 32] =
        (_Float16)(vals[i] * inv * __expf(zmin - zl[i]));
  if (lane == 0) rsc[row] = __expf(32.0f - zmin);
}

// ------------- stage B: ov = (rowscale * (QZ @ ES^T), causal) @ V
__global__ __launch_bounds__(256) void attn2_kernel(
    const _Float16* __restrict__ qzh, const _Float16* __restrict__ es,
    const _Float16* __restrict__ vh, const float* __restrict__ rsc,
    float* __restrict__ ovbuf) {
  const int qt   = blockIdx.x;
  const int bh   = blockIdx.y;
  const int lane = threadIdx.x & 31;
  const int wave = threadIdx.x >> 5;
  __shared__ _Float16 ldsW[16][40];    // masked/scaled W tile [t][tau]
  __shared__ _Float16 sVT[256][40];    // V chunk transposed [dv][tau]

  const _Float16* qzbase = qzh + ((size_t)bh * T_ + qt * 16) * M_;
  const _Float16* ebase  = es + (size_t)bh * T_ * M_;
  const _Float16* vbase  = vh + (size_t)bh * T_ * DV_;
  const float*    rbase  = rsc + (size_t)bh * T_;
  const int n0 = wave * 32;

  v16h aq[8];
  if (wave < 2) {
#pragma unroll
    for (int mc = 0; mc < 8; ++mc)
      aq[mc] = load_frag_rowmajor(qzbase + mc * 32, M_, lane);
  }

  const v8f vzero = {0.f, 0.f, 0.f, 0.f, 0.f, 0.f, 0.f, 0.f};
  v8f accO[2] = {vzero, vzero};

  const int nchunks = (qt >> 1) + 1;
  for (int c = 0; c < nchunks; ++c) {
    const int t0 = c * 32;
    // stage V chunk (32 tau x 256 dv) transposed
#pragma unroll
    for (int u = 0; u < 4; ++u) {
      const int g   = threadIdx.x + u * 256;
      const int tau = g >> 5;
      const int vg  = (g & 31) * 8;
      const uint4 d = *reinterpret_cast<const uint4*>(
          &vbase[(size_t)(t0 + tau) * DV_ + vg]);
      const _Float16* hp = reinterpret_cast<const _Float16*>(&d);
#pragma unroll
      for (int j = 0; j < 8; ++j) sVT[vg + j][tau] = hp[j];
    }
    if (wave < 2) {  // W'[t,tau] sub-tile, contraction over M
      v8f accW = vzero;
      const _Float16* eb = ebase + (size_t)(t0 + wave * 16) * M_;
#pragma unroll
      for (int mc = 0; mc < 8; ++mc) {
        v16h bf = load_frag_rowmajor(eb + mc * 32, M_, lane);  // ES^T gather
        accW = wmma_f16(aq[mc], bf, accW);
      }
      const int col  = lane & 15;
      const int rofs = (lane >> 4) << 3;
      const int taug = t0 + wave * 16 + col;
#pragma unroll
      for (int i = 0; i < 8; ++i) {
        const int row = i + rofs;
        const int tg  = qt * 16 + row;
        const float w = (taug <= tg) ? accW[i] * rbase[tg] : 0.0f;
        ldsW[row][wave * 16 + col] = (_Float16)w;  // W in [0,1]
      }
    }
    __syncthreads();
    v16h af = load_frag_rowmajor(&ldsW[0][0], 40, lane);
#pragma unroll
    for (int s = 0; s < 2; ++s) {
      v16h bf = load_frag_rowmajor(&sVT[n0 + s * 16][0], 40, lane);
      accO[s] = wmma_f16(af, bf, accO[s]);
    }
    __syncthreads();
  }
  const int col  = lane & 15;
  const int rofs = (lane >> 4) << 3;
#pragma unroll
  for (int s = 0; s < 2; ++s)
#pragma unroll
    for (int i = 0; i < 8; ++i) {
      const int t = qt * 16 + i + rofs;
      ovbuf[((size_t)bh * T_ + t) * DV_ + n0 + s * 16 + col] = accO[s][i];
    }
}

// ------------------ RMS-norm * gn * g * sigmoid(g) -> f16 for output GEMM
__global__ __launch_bounds__(256) void rmsgate_kernel(
    const float* __restrict__ ovbuf, const float* __restrict__ Y5,
    const float* __restrict__ gnw, _Float16* __restrict__ oh) {
  const int lane = threadIdx.x & 31;
  const int wave = threadIdx.x >> 5;
  const size_t row = (size_t)blockIdx.x * 8 + wave;  // bh*T + t
  const int t  = (int)(row % T_);
  const int bh = (int)(row / T_);
  const int h  = bh & 3, b = bh >> 2;
  const float* o = ovbuf + row * DV_;
  float ov[8], ss = 0.f;
#pragma unroll
  for (int i = 0; i < 8; ++i) {
    ov[i] = o[lane + 32 * i];
    ss += ov[i] * ov[i];
  }
#pragma unroll
  for (int off = 16; off; off >>= 1) ss += __shfl_xor(ss, off, 32);
  const float r = rsqrtf(ss * (1.0f / DV_) + 1e-5f);
  const size_t grow = (size_t)(b * T_ + t) * N5_ + 3 * D_ + h * DV_;
  const size_t drow = (size_t)(b * T_ + t) * D_ + h * DV_;
#pragma unroll
  for (int i = 0; i < 8; ++i) {
    const int vv = lane + 32 * i;
    const float g   = Y5[grow + vv];
    const float sig = 1.0f / (1.0f + __expf(-g));
    oh[drow + vv] = (_Float16)(ov[i] * r * gnw[vv] * g * sig);
  }
}

// ---------------------------------------------------------------- launcher
extern "C" void kernel_launch(void* const* d_in, const int* in_sizes, int n_in,
                              void* d_out, int out_size, void* d_ws,
                              size_t ws_size, hipStream_t stream) {
  (void)in_sizes; (void)n_in; (void)out_size; (void)ws_size;
  const float* x   = (const float*)d_in[0];
  const float* qw  = (const float*)d_in[1];
  const float* kw  = (const float*)d_in[2];
  const float* vw  = (const float*)d_in[3];
  const float* gw  = (const float*)d_in[4];
  const float* sw  = (const float*)d_in[5];
  const float* sgw = (const float*)d_in[6];
  const float* gnw = (const float*)d_in[7];
  const float* ow  = (const float*)d_in[8];
  float* out = (float*)d_out;

  char* p = (char*)d_ws;
  auto take = [&](size_t bytes) {
    char* r = p;
    p += (bytes + 255) & ~(size_t)255;
    return r;
  };
  _Float16* xh  = (_Float16*)take((size_t)B_ * T_ * D_ * 2);
  _Float16* wh  = (_Float16*)take((size_t)D_ * NW_ * 2);
  float*    Y5  = (float*)   take((size_t)B_ * T_ * N5_ * 4);
  float*    sgb = (float*)   take((size_t)B_ * H_ * T_ * 4);
  _Float16* qh  = (_Float16*)take((size_t)B_ * H_ * T_ * DK_ * 2);
  _Float16* kh  = (_Float16*)take((size_t)B_ * H_ * T_ * DK_ * 2);
  _Float16* vh  = (_Float16*)take((size_t)B_ * H_ * T_ * DV_ * 2);
  _Float16* es  = (_Float16*)take((size_t)B_ * H_ * T_ * M_ * 2);
  float*    zb  = (float*)   take((size_t)B_ * H_ * T_ * M_ * 4);
  float*    okb = (float*)   take((size_t)B_ * H_ * T_ * M_ * 4);
  _Float16* qzh = (_Float16*)take((size_t)B_ * H_ * T_ * M_ * 2);
  float*    rsc = (float*)   take((size_t)B_ * H_ * T_ * 4);
  float*    ovb = (float*)   take((size_t)B_ * H_ * T_ * DV_ * 4);
  _Float16* oh  = (_Float16*)take((size_t)B_ * T_ * D_ * 2);

  // casts
  cast_x_kernel<<<(B_ * T_ * D_) / 256, 256, 0, stream>>>(x, xh);
  cast_w_kernel<<<(D_ * NW_) / 256, 256, 0, stream>>>(qw, kw, vw, gw, sw, ow, wh);

  // fused q|k|v|g|s projection: (8192x1024) x (1024x5120)
  gemm_kernel<<<dim3((B_ * T_) / 128, N5_ / 128), 256, 0, stream>>>(
      xh, wh, Y5, D_, NW_, N5_, B_ * T_);

  // gate chain
  sg_kernel<<<(B_ * T_ * H_) / 256, 256, 0, stream>>>(x, sgw, sgb);
  cumsum_kernel<<<1, 32, 0, stream>>>(sgb);
  rope_kernel<<<(B_ * H_ * T_ * 128) / 256, 256, 0, stream>>>(Y5, sgb, qh, kh);
  zscan_kernel<<<(B_ * H_ * M_) / 256, 256, 0, stream>>>(Y5, zb, es);
  vcast_kernel<<<(B_ * H_ * T_ * DV_) / 256, 256, 0, stream>>>(Y5, vh);

  // chunked-causal linear attention, two WMMA stages
  attn1_kernel<<<dim3(T_ / 16, B_ * H_), 256, 0, stream>>>(qh, kh, es, zb, okb);
  softmax_qz_kernel<<<(B_ * H_ * T_) / 8, 256, 0, stream>>>(okb, zb, qzh, rsc);
  attn2_kernel<<<dim3(T_ / 16, B_ * H_), 256, 0, stream>>>(qzh, es, vh, rsc, ovb);

  // epilogue + output projection
  rmsgate_kernel<<<(B_ * H_ * T_) / 8, 256, 0, stream>>>(ovb, Y5, gnw, oh);
  gemm_kernel<<<dim3((B_ * T_) / 128, D_ / 128), 256, 0, stream>>>(
      oh, wh + N5_, out, D_, NW_, D_, B_ * T_);
}
